// ConvAttention_10101763080855
// MI455X (gfx1250) — matrix operand order repaired and checked by reading, not compiled, not run on hardware
//
#include <hip/hip_runtime.h>

#define BATCH  8
#define CCH    512
#define NTOK   4096      // H*W = 64*64
#define GROUPS 32
#define CPG    16        // channels per group
#define GELEM  (CPG*NTOK)
#define EPSV   1e-6f

typedef __attribute__((ext_vector_type(16))) __bf16       v16bf;
typedef __attribute__((ext_vector_type(8)))  float        v8f;
typedef __attribute__((ext_vector_type(4)))  unsigned int u32x4;
typedef __attribute__((ext_vector_type(8)))  unsigned int u32x8;
typedef __attribute__((ext_vector_type(4)))  float        f32x4;

// ---------- helpers ----------
__device__ __forceinline__ unsigned short f2bf(float f) {
    unsigned int u = __builtin_bit_cast(unsigned int, f);
    u = (u + 0x7FFFu + ((u >> 16) & 1u)) >> 16;
    return (unsigned short)u;
}

__device__ __forceinline__ v16bf pack16(u32x4 a, u32x4 b) {
    u32x8 v;
    v[0]=a[0]; v[1]=a[1]; v[2]=a[2]; v[3]=a[3];
    v[4]=b[0]; v[5]=b[1]; v[6]=b[2]; v[7]=b[3];
    return __builtin_bit_cast(v16bf, v);
}

// A fragment 16x32 bf16 (MxK), source layout [row][K] row-major, ld in elements.
// lane m=l%16, hi=l/16; VGPR0-3 hold K = k0+8*hi .. +7 ; VGPR4-7 hold K = k0+16+8*hi .. +7
__device__ __forceinline__ v16bf load_frag_a(const unsigned short* base, int ld,
                                             int row0, int k0, int lane) {
    int m = lane & 15, hi = (lane >> 4) & 1;
    const unsigned short* p = base + (size_t)(row0 + m) * ld + k0 + 8*hi;
    return pack16(*(const u32x4*)p, *(const u32x4*)(p + 16));
}

// B fragment 32x16 bf16 (KxN), source layout [col][K] row-major.
// lane n=l%16, hi=l/16; VGPR0-3 hold K = k0+8*hi .. +7 ; VGPR4-7 hold K = k0+16+8*hi .. +7
__device__ __forceinline__ v16bf load_frag_b(const unsigned short* base, int ld,
                                             int col0, int k0, int lane) {
    int n = lane & 15, hi = (lane >> 4) & 1;
    const unsigned short* p = base + (size_t)(col0 + n) * ld + k0 + 8*hi;
    return pack16(*(const u32x4*)p, *(const u32x4*)(p + 16));
}

__device__ __forceinline__ v8f wmma_bf16(v16bf a, v16bf b, v8f c) {
    return __builtin_amdgcn_wmma_f32_16x16x32_bf16(false, a, false, b,
                                                   (short)0, c, false, false);
}

// ---------- kernel 1: group-norm statistics ----------
__global__ void k_gn_stats(const float* __restrict__ x, float* __restrict__ stats) {
    int bg = blockIdx.x;                       // b*GROUPS + g, contiguous 65536 floats
    const f32x4* p = (const f32x4*)(x + (size_t)bg * GELEM);
    float s = 0.f, ss = 0.f;
    for (int i = threadIdx.x; i < GELEM/4; i += 256) {
        f32x4 v = p[i];
        s  += v[0]+v[1]+v[2]+v[3];
        ss += v[0]*v[0]+v[1]*v[1]+v[2]*v[2]+v[3]*v[3];
    }
    __shared__ float rs[256], rss[256];
    rs[threadIdx.x] = s; rss[threadIdx.x] = ss;
    __syncthreads();
    for (int o = 128; o > 0; o >>= 1) {
        if (threadIdx.x < o) { rs[threadIdx.x] += rs[threadIdx.x+o];
                               rss[threadIdx.x] += rss[threadIdx.x+o]; }
        __syncthreads();
    }
    if (threadIdx.x == 0) {
        float mean = rs[0] * (1.f/GELEM);
        float var  = rss[0] * (1.f/GELEM) - mean*mean;
        stats[2*bg]   = mean;
        stats[2*bg+1] = rsqrtf(var + EPSV);
    }
}

// ---------- kernel 2: apply GN, convert to bf16, transpose to gT (B,N,C) ----------
__global__ void k_gn_apply_t(const float* __restrict__ x, const float* __restrict__ stats,
                             const float* __restrict__ gw, const float* __restrict__ gb,
                             unsigned short* __restrict__ gT) {
    int n0 = blockIdx.x * 64, c0 = blockIdx.y * 32, b = blockIdx.z;
    __shared__ unsigned short tile[64][34];          // [n][c] (padded)
    int t = threadIdx.x;
    #pragma unroll
    for (int j = 0; j < 2; ++j) {
        int idx  = t + 256*j;                        // 512 float4 units
        int row  = idx >> 4;                         // c offset 0..31
        int coln = (idx & 15) << 2;                  // n offset 0..60
        int c = c0 + row;
        float mean = stats[2*(b*GROUPS + (c >> 4))];
        float rstd = stats[2*(b*GROUPS + (c >> 4)) + 1];
        float sw = gw[c] * rstd;
        float sb = gb[c] - mean * sw;
        f32x4 v = *(const f32x4*)(x + (size_t)b*CCH*NTOK + (size_t)c*NTOK + n0 + coln);
        #pragma unroll
        for (int q = 0; q < 4; ++q)
            tile[coln + q][row] = f2bf(v[q]*sw + sb);
    }
    __syncthreads();
    int n = t >> 2, off = (t & 3) << 3;              // 8 bf16 = 16B per thread
    u32x4 o;
    #pragma unroll
    for (int q = 0; q < 4; ++q)
        o[q] = (unsigned int)tile[n][off+2*q] | ((unsigned int)tile[n][off+2*q+1] << 16);
    *(u32x4*)(gT + (size_t)b*NTOK*CCH + (size_t)(n0+n)*CCH + c0 + off) = o;
}

// ---------- kernel 3: convert weights fp32 -> bf16 ----------
__global__ void k_cvt_w(const float* __restrict__ wq, const float* __restrict__ wk,
                        const float* __restrict__ wv, unsigned short* __restrict__ wb) {
    int i = blockIdx.x * 256 + threadIdx.x;          // float4 index over 3*C*C
    const int per = (CCH*CCH) >> 2;
    const float* src = (i < per) ? wq : (i < 2*per ? wk : wv);
    int li = i % per;
    f32x4 v = ((const f32x4*)src)[li];
    unsigned short* dst = wb + (size_t)(i/per)*CCH*CCH + (size_t)li*4;
    dst[0]=f2bf(v[0]); dst[1]=f2bf(v[1]); dst[2]=f2bf(v[2]); dst[3]=f2bf(v[3]);
}

// ---------- kernel 4: QKV GEMM  out = W(C,C) @ g(C,N) + bias ----------
// q,k stored (C,N) bf16; v stored transposed (N,C) bf16
__global__ void k_qkv(const unsigned short* __restrict__ wb,
                      const float* __restrict__ bq, const float* __restrict__ bk,
                      const float* __restrict__ bv,
                      const unsigned short* __restrict__ gT,
                      unsigned short* __restrict__ qb, unsigned short* __restrict__ kb,
                      unsigned short* __restrict__ vT) {
    int n0   = blockIdx.x * 16;
    int wave = threadIdx.x >> 5, lane = threadIdx.x & 31;
    int m0   = blockIdx.y * 128 + wave * 16;
    int b    = blockIdx.z / 3, sel = blockIdx.z % 3;
    const unsigned short* W = wb + (size_t)sel * CCH * CCH;
    const unsigned short* g = gT + (size_t)b * NTOK * CCH;
    const float* bias = sel == 0 ? bq : (sel == 1 ? bk : bv);
    __shared__ unsigned short vt[16][128];
    v8f acc = {0,0,0,0,0,0,0,0};
    for (int k0 = 0; k0 < CCH; k0 += 32) {
        v16bf a  = load_frag_a(W, CCH, m0, k0, lane);
        v16bf bf = load_frag_b(g, CCH, n0, k0, lane);
        acc = wmma_bf16(a, bf, acc);
    }
    int hl = lane >> 4, nl = lane & 15;
    if (sel < 2) {
        unsigned short* dst = (sel == 0 ? qb : kb) + (size_t)b * CCH * NTOK;
        #pragma unroll
        for (int r = 0; r < 8; ++r) {
            int m = m0 + r + 8*hl;
            dst[(size_t)m * NTOK + n0 + nl] = f2bf(acc[r] + bias[m]);
        }
    } else {
        #pragma unroll
        for (int r = 0; r < 8; ++r) {
            int m = m0 + r + 8*hl;
            vt[nl][wave*16 + r + 8*hl] = f2bf(acc[r] + bias[m]);
        }
        __syncthreads();
        int t = threadIdx.x;
        int n = t >> 4, off = (t & 15) << 3;
        u32x4 o;
        #pragma unroll
        for (int q = 0; q < 4; ++q)
            o[q] = (unsigned int)vt[n][off+2*q] | ((unsigned int)vt[n][off+2*q+1] << 16);
        *(u32x4*)(vT + (size_t)b*NTOK*CCH + (size_t)(n0+n)*CCH + blockIdx.y*128 + off) = o;
    }
}

// ---------- kernel 5: scores = (q k^T)*C^-0.5, fused row-softmax -> probs bf16 ----------
__global__ void k_scores_softmax(const unsigned short* __restrict__ qb,
                                 const unsigned short* __restrict__ kb,
                                 unsigned short* __restrict__ probs) {
    int cq0 = blockIdx.x * 16, b = blockIdx.y;
    int wave = threadIdx.x >> 5, lane = threadIdx.x & 31;
    const unsigned short* q = qb + (size_t)b * CCH * NTOK;
    const unsigned short* k = kb + (size_t)b * CCH * NTOK;
    int ck0 = wave * 64;
    v8f a0={0,0,0,0,0,0,0,0}, a1=a0, a2=a0, a3=a0;
    for (int k0 = 0; k0 < NTOK; k0 += 32) {
        v16bf af = load_frag_a(q, NTOK, cq0, k0, lane);
        a0 = wmma_bf16(af, load_frag_b(k, NTOK, ck0,      k0, lane), a0);
        a1 = wmma_bf16(af, load_frag_b(k, NTOK, ck0 + 16, k0, lane), a1);
        a2 = wmma_bf16(af, load_frag_b(k, NTOK, ck0 + 32, k0, lane), a2);
        a3 = wmma_bf16(af, load_frag_b(k, NTOK, ck0 + 48, k0, lane), a3);
    }
    __shared__ float sc[16][CCH + 4];
    __shared__ float red[16][16];
    __shared__ float rowred[16];
    const float scale = 0.044194173824159216f;       // 512^-0.5
    int hl = lane >> 4, nl = lane & 15;
    #pragma unroll
    for (int r = 0; r < 8; ++r) {
        int row = r + 8*hl;
        sc[row][ck0      + nl] = a0[r] * scale;
        sc[row][ck0 + 16 + nl] = a1[r] * scale;
        sc[row][ck0 + 32 + nl] = a2[r] * scale;
        sc[row][ck0 + 48 + nl] = a3[r] * scale;
    }
    __syncthreads();
    int row = threadIdx.x >> 4, j = threadIdx.x & 15;   // 16 threads per row
    float mx = -1e30f;
    for (int i = 0; i < 32; ++i) mx = fmaxf(mx, sc[row][j + 16*i]);
    red[row][j] = mx; __syncthreads();
    if (j == 0) { float m2 = red[row][0];
                  for (int i = 1; i < 16; ++i) m2 = fmaxf(m2, red[row][i]);
                  rowred[row] = m2; }
    __syncthreads();
    float rm = rowred[row];
    float s = 0.f;
    for (int i = 0; i < 32; ++i) {
        float e = __expf(sc[row][j + 16*i] - rm);
        sc[row][j + 16*i] = e; s += e;
    }
    red[row][j] = s; __syncthreads();
    if (j == 0) { float t2 = 0.f;
                  for (int i = 0; i < 16; ++i) t2 += red[row][i];
                  rowred[row] = t2; }
    __syncthreads();
    float inv = 1.f / rowred[row];
    unsigned short* pr = probs + (size_t)b*CCH*CCH + (size_t)(cq0+row)*CCH;
    for (int i = 0; i < 32; ++i) pr[j + 16*i] = f2bf(sc[row][j + 16*i] * inv);
}

// ---------- kernel 6: attn = probs @ v, permuted residual store ----------
// out[b].flat[n*C + c] = x[b].flat[n*C + c] + attn[b][c][n]
__global__ void k_attn_out(const unsigned short* __restrict__ probs,
                           const unsigned short* __restrict__ vT,
                           const float* __restrict__ x, float* __restrict__ out) {
    int n0 = blockIdx.x * 16, b = blockIdx.y;
    int wave = threadIdx.x >> 5, lane = threadIdx.x & 31;
    const unsigned short* P = probs + (size_t)b * CCH * CCH;
    const unsigned short* V = vT    + (size_t)b * NTOK * CCH;
    int c0 = wave * 64;
    v8f a0={0,0,0,0,0,0,0,0}, a1=a0, a2=a0, a3=a0;
    for (int k0 = 0; k0 < CCH; k0 += 32) {
        v16bf bf = load_frag_b(V, CCH, n0, k0, lane);
        a0 = wmma_bf16(load_frag_a(P, CCH, c0,      k0, lane), bf, a0);
        a1 = wmma_bf16(load_frag_a(P, CCH, c0 + 16, k0, lane), bf, a1);
        a2 = wmma_bf16(load_frag_a(P, CCH, c0 + 32, k0, lane), bf, a2);
        a3 = wmma_bf16(load_frag_a(P, CCH, c0 + 48, k0, lane), bf, a3);
    }
    __shared__ float at[16][CCH + 4];                // [n_local][c]
    int hl = lane >> 4, nl = lane & 15;
    #pragma unroll
    for (int r = 0; r < 8; ++r) {
        int cl = r + 8*hl;
        at[nl][c0      + cl] = a0[r];
        at[nl][c0 + 16 + cl] = a1[r];
        at[nl][c0 + 32 + cl] = a2[r];
        at[nl][c0 + 48 + cl] = a3[r];
    }
    __syncthreads();
    const float* xb = x   + (size_t)b*CCH*NTOK;
    float*       ob = out + (size_t)b*CCH*NTOK;
    #pragma unroll
    for (int i = 0; i < 8; ++i) {
        int idx  = threadIdx.x + 256*i;              // float4 units, 2048 total
        int n    = idx >> 5;                         // 32 float4 per 512-float row
        int coff = (idx & 31) << 2;
        size_t o = (size_t)(n0+n)*CCH + coff;
        f32x4 xv = *(const f32x4*)(xb + o);
        f32x4 ov;
        ov[0]=xv[0]+at[n][coff];   ov[1]=xv[1]+at[n][coff+1];
        ov[2]=xv[2]+at[n][coff+2]; ov[3]=xv[3]+at[n][coff+3];
        *(f32x4*)(ob + o) = ov;
    }
}

// ---------- launch ----------
extern "C" void kernel_launch(void* const* d_in, const int* in_sizes, int n_in,
                              void* d_out, int out_size, void* d_ws, size_t ws_size,
                              hipStream_t stream) {
    (void)in_sizes; (void)n_in; (void)out_size; (void)ws_size;
    const float* x   = (const float*)d_in[0];
    const float* gnw = (const float*)d_in[1];
    const float* gnb = (const float*)d_in[2];
    const float* wq  = (const float*)d_in[3];
    const float* bq  = (const float*)d_in[4];
    const float* wk  = (const float*)d_in[5];
    const float* bk  = (const float*)d_in[6];
    const float* wv  = (const float*)d_in[7];
    const float* bv  = (const float*)d_in[8];
    float* out = (float*)d_out;

    char* ws = (char*)d_ws;
    const size_t SZ_W  = (size_t)3*CCH*CCH*2;            // 1.5 MB bf16 weights
    const size_t SZ_BT = (size_t)BATCH*NTOK*CCH*2;       // 32 MB per bf16 tensor
    float*          stats = (float*)ws;                          // 2 KB
    unsigned short* wb    = (unsigned short*)(ws + 4096);
    unsigned short* gT    = (unsigned short*)(ws + 4096 + SZ_W);
    unsigned short* qb    = (unsigned short*)(ws + 4096 + SZ_W + SZ_BT);
    unsigned short* kb    = (unsigned short*)(ws + 4096 + SZ_W + 2*SZ_BT);
    unsigned short* vT    = (unsigned short*)(ws + 4096 + SZ_W + 3*SZ_BT);
    unsigned short* probs = (unsigned short*)(ws + 4096 + SZ_W + 4*SZ_BT);  // 4 MB

    k_gn_stats      <<<BATCH*GROUPS, 256, 0, stream>>>(x, stats);
    k_gn_apply_t    <<<dim3(NTOK/64, CCH/32, BATCH), 256, 0, stream>>>(x, stats, gnw, gnb, gT);
    k_cvt_w         <<<(3*CCH*CCH/4)/256, 256, 0, stream>>>(wq, wk, wv, wb);
    k_qkv           <<<dim3(NTOK/16, CCH/128, BATCH*3), 256, 0, stream>>>(wb, bq, bk, bv, gT, qb, kb, vT);
    k_scores_softmax<<<dim3(CCH/16, BATCH), 256, 0, stream>>>(qb, kb, probs);
    k_attn_out      <<<dim3(NTOK/16, BATCH), 256, 0, stream>>>(probs, vT, x, out);
}
